// LocalAttention_68212670595412
// MI455X (gfx1250) — compile-verified, matrix operand order checked by reading, not run on hardware
//
#include <hip/hip_runtime.h>
#include <hip/hip_bf16.h>

typedef __attribute__((ext_vector_type(16))) __bf16 v16bf;
typedef __attribute__((ext_vector_type(8)))  float  v8f;

#define N_TOK 4096
#define C_DIM 768
#define HEADS 12
#define HD    64
#define WIN   128

__device__ __forceinline__ unsigned short f2bf(float f) {
  unsigned int u = __float_as_uint(f);
  unsigned int r = (u + 0x7FFFu + ((u >> 16) & 1u)) >> 16;
  return (unsigned short)r;
}

union FragU { v16bf v; uint4 q[2]; };

// Load a 16x32 bf16 A-fragment (or a 32x16 B-fragment whose logical (k,n)
// element lives at stored[n][k] row-major). Per-lane: row = row0 + lane%16,
// k-chunks of 8 bf16 at k0 + (lane/16)*8 and +16 (two 16B loads).
__device__ __forceinline__ v16bf load_frag(const unsigned short* base, int row0,
                                           int ld, int k0) {
  int lane = threadIdx.x & 31;
  int r = row0 + (lane & 15);
  int k = k0 + ((lane >> 4) << 3);
  const uint4* p = (const uint4*)(base + (size_t)r * ld + k);
  FragU f;
  f.q[0] = p[0];   // elements k .. k+7
  f.q[1] = p[2];   // elements k+16 .. k+23
  return f.v;
}

__device__ __forceinline__ v8f wmma_bf16(v16bf a, v16bf b, v8f c) {
  return __builtin_amdgcn_wmma_f32_16x16x32_bf16(false, a, false, b,
                                                 (short)0, c, false, false);
}

__global__ void cvt_f32_bf16(const float* __restrict__ in,
                             unsigned short* __restrict__ out, int n) {
  int i = blockIdx.x * blockDim.x + threadIdx.x;
  if (i < n) out[i] = f2bf(in[i]);
}

// ---- QKV projection: y = x @ Wqkv^T + b, scatter q/k/vT ----
// Per wave: 32 rows x 64 cols output (2 A-frags x 4 B-frags -> 8 WMMA/k-step)
// launch_bounds(...,1): allow full VGPR budget, no accumulator spills.
__global__ void __launch_bounds__(128, 1)
qkv_gemm(const unsigned short* __restrict__ xb,
         const unsigned short* __restrict__ wq,
         const float* __restrict__ bias,
         unsigned short* __restrict__ qb,
         unsigned short* __restrict__ kb,
         unsigned short* __restrict__ vtb) {
  int wave = threadIdx.x >> 5;
  int lane = threadIdx.x & 31;
  int row0 = (blockIdx.x * 4 + wave) * 32;
  int col0 = blockIdx.y * 64;
  v8f acc[2][4] = {};
  for (int kc = 0; kc < C_DIM; kc += 32) {
    v16bf a0 = load_frag(xb, row0,      C_DIM, kc);
    v16bf a1 = load_frag(xb, row0 + 16, C_DIM, kc);
#pragma unroll
    for (int t = 0; t < 4; ++t) {
      v16bf b = load_frag(wq, col0 + t * 16, C_DIM, kc);
      acc[0][t] = wmma_bf16(a0, b, acc[0][t]);
      acc[1][t] = wmma_bf16(a1, b, acc[1][t]);
    }
  }
  int mbase = (lane >> 4) << 3;
#pragma unroll
  for (int m2 = 0; m2 < 2; ++m2) {
#pragma unroll
    for (int t = 0; t < 4; ++t) {
      int o = col0 + t * 16 + (lane & 15);
      float bv = bias[o];
      int which = o / C_DIM;
      int rem = o - which * C_DIM;
      int h = rem >> 6;
      int d = rem & 63;
#pragma unroll
      for (int r = 0; r < 8; ++r) {
        int n = row0 + m2 * 16 + mbase + r;
        float val = acc[m2][t][r] + bv;
        if (which == 0)       qb[((size_t)h * N_TOK + n) * HD + d] = f2bf(val * 0.125f);
        else if (which == 1)  kb[((size_t)h * N_TOK + n) * HD + d] = f2bf(val);
        else                  vtb[((size_t)h * HD + d) * N_TOK + n] = f2bf(val);
      }
    }
  }
}

// ---------------- Banded flash attention, one wave = 16 query rows ----------
__global__ void local_attn(const unsigned short* __restrict__ qb,
                           const unsigned short* __restrict__ kb,
                           const unsigned short* __restrict__ vtb,
                           unsigned short* __restrict__ ab) {
  __shared__ __align__(16) unsigned short pbuf[4][16 * 32];
  int wave = threadIdx.x >> 5;
  int lane = threadIdx.x & 31;
  int wid  = blockIdx.x * 4 + wave;
  int h    = wid >> 8;       // 256 row tiles per head
  int tile = wid & 255;
  int i0   = tile * 16;
  const unsigned short* qh = qb  + (size_t)h * N_TOK * HD;
  const unsigned short* kh = kb  + (size_t)h * N_TOK * HD;
  const unsigned short* vh = vtb + (size_t)h * HD * N_TOK;

  v16bf qa0 = load_frag(qh, i0, HD, 0);
  v16bf qa1 = load_frag(qh, i0, HD, 32);

  v8f o0 = {}, o1 = {}, o2 = {}, o3 = {};
  float mrow[8], lrow[8];
#pragma unroll
  for (int r = 0; r < 8; ++r) { mrow[r] = -1e30f; lrow[r] = 0.f; }

  int j_lo = i0 - WIN; if (j_lo < 0) j_lo = 0;
  int j_hi = i0 + 15 + WIN; if (j_hi > N_TOK - 1) j_hi = N_TOK - 1;
  int mbase = (lane >> 4) << 3;
  int ncol  = lane & 15;
  int kchunk = (lane >> 4) << 3;

  for (int jb = j_lo; jb <= j_hi; jb += 32) {
    // ---- S = Q @ K^T for two 16-column subtiles
    v8f s[2];
#pragma unroll
    for (int sub = 0; sub < 2; ++sub) {
      int jr = jb + sub * 16 + ncol;
      if (jr > N_TOK - 1) jr = N_TOK - 1;               // clamp (masked later)
      const uint4* p = (const uint4*)(kh + (size_t)jr * HD + kchunk);
      FragU b0, b1;
      b0.q[0] = p[0]; b0.q[1] = p[2];                   // dims 0..31
      b1.q[0] = p[4]; b1.q[1] = p[6];                   // dims 32..63
      v8f c = {};
      c = wmma_bf16(qa0, b0.v, c);
      c = wmma_bf16(qa1, b1.v, c);
      s[sub] = c;
    }
    // ---- band mask + row max over 32 columns
    float p0[8], p1[8], mx[8];
#pragma unroll
    for (int r = 0; r < 8; ++r) {
      int i  = i0 + mbase + r;
      int j0 = jb + ncol;
      int j1 = jb + 16 + ncol;
      bool ok0 = (j0 >= i - WIN) && (j0 <= i + WIN) && (j0 < N_TOK);
      bool ok1 = (j1 >= i - WIN) && (j1 <= i + WIN) && (j1 < N_TOK);
      p0[r] = ok0 ? s[0][r] : -1e30f;
      p1[r] = ok1 ? s[1][r] : -1e30f;
      mx[r] = fmaxf(p0[r], p1[r]);
    }
#pragma unroll
    for (int off = 1; off < 16; off <<= 1)
#pragma unroll
      for (int r = 0; r < 8; ++r)
        mx[r] = fmaxf(mx[r], __shfl_xor(mx[r], off, 32));
    // ---- online softmax update
    float alpha[8];
#pragma unroll
    for (int r = 0; r < 8; ++r) {
      float mn = fmaxf(mrow[r], mx[r]);
      alpha[r] = __expf(mrow[r] - mn);
      mrow[r]  = mn;
      p0[r] = __expf(p0[r] - mn);
      p1[r] = __expf(p1[r] - mn);
    }
    float rs[8];
#pragma unroll
    for (int r = 0; r < 8; ++r) rs[r] = p0[r] + p1[r];
#pragma unroll
    for (int off = 1; off < 16; off <<= 1)
#pragma unroll
      for (int r = 0; r < 8; ++r) rs[r] += __shfl_xor(rs[r], off, 32);
#pragma unroll
    for (int r = 0; r < 8; ++r) {
      lrow[r] = lrow[r] * alpha[r] + rs[r];
      o0[r] *= alpha[r]; o1[r] *= alpha[r];
      o2[r] *= alpha[r]; o3[r] *= alpha[r];
    }
    // ---- transpose P (C-layout -> A-layout) through LDS, as bf16
    unsigned short* pb = pbuf[wave];
#pragma unroll
    for (int r = 0; r < 8; ++r) {
      int m = mbase + r;
      pb[m * 32 + ncol]      = f2bf(p0[r]);
      pb[m * 32 + 16 + ncol] = f2bf(p1[r]);
    }
    asm volatile("s_wait_dscnt 0" ::: "memory");
    FragU pa;
    const uint4* pp = (const uint4*)(pb + (lane & 15) * 32 + kchunk);
    pa.q[0] = pp[0]; pa.q[1] = pp[2];
    // ---- O += P @ V  (V stored transposed [64][N])
    int k1 = jb + kchunk;       if (k1 > N_TOK - 8) k1 = N_TOK - 8;
    int k2 = jb + 16 + kchunk;  if (k2 > N_TOK - 8) k2 = N_TOK - 8;
#pragma unroll
    for (int t = 0; t < 4; ++t) {
      const unsigned short* vrow = vh + (size_t)(t * 16 + ncol) * N_TOK;
      FragU vb;
      vb.q[0] = *(const uint4*)(vrow + k1);
      vb.q[1] = *(const uint4*)(vrow + k2);
      v8f& acc = (t == 0) ? o0 : (t == 1) ? o1 : (t == 2) ? o2 : o3;
      acc = wmma_bf16(pa.v, vb.v, acc);
    }
  }
  // ---- normalize and emit bf16 [N][C] with c = h*64 + d
  float inv[8];
#pragma unroll
  for (int r = 0; r < 8; ++r) inv[r] = 1.0f / lrow[r];
#pragma unroll
  for (int t = 0; t < 4; ++t) {
    v8f& acc = (t == 0) ? o0 : (t == 1) ? o1 : (t == 2) ? o2 : o3;
#pragma unroll
    for (int r = 0; r < 8; ++r) {
      int n = i0 + mbase + r;
      ab[(size_t)n * C_DIM + h * HD + t * 16 + ncol] = f2bf(acc[r] * inv[r]);
    }
  }
}

// ---- Output projection: out = attn @ Wproj^T + b (f32) ----
// Per wave: 32 rows x 64 cols output
__global__ void __launch_bounds__(128, 1)
proj_gemm(const unsigned short* __restrict__ ab,
          const unsigned short* __restrict__ wp,
          const float* __restrict__ bias,
          float* __restrict__ out) {
  int wave = threadIdx.x >> 5;
  int lane = threadIdx.x & 31;
  int row0 = (blockIdx.x * 4 + wave) * 32;
  int col0 = blockIdx.y * 64;
  v8f acc[2][4] = {};
  for (int kc = 0; kc < C_DIM; kc += 32) {
    v16bf a0 = load_frag(ab, row0,      C_DIM, kc);
    v16bf a1 = load_frag(ab, row0 + 16, C_DIM, kc);
#pragma unroll
    for (int t = 0; t < 4; ++t) {
      v16bf b = load_frag(wp, col0 + t * 16, C_DIM, kc);
      acc[0][t] = wmma_bf16(a0, b, acc[0][t]);
      acc[1][t] = wmma_bf16(a1, b, acc[1][t]);
    }
  }
  int mbase = (lane >> 4) << 3;
#pragma unroll
  for (int m2 = 0; m2 < 2; ++m2) {
#pragma unroll
    for (int t = 0; t < 4; ++t) {
      int o = col0 + t * 16 + (lane & 15);
      float bv = bias[o];
#pragma unroll
      for (int r = 0; r < 8; ++r)
        out[(size_t)(row0 + m2 * 16 + mbase + r) * C_DIM + o] = acc[m2][t][r] + bv;
    }
  }
}

extern "C" void kernel_launch(void* const* d_in, const int* in_sizes, int n_in,
                              void* d_out, int out_size, void* d_ws, size_t ws_size,
                              hipStream_t stream) {
  (void)in_sizes; (void)n_in; (void)out_size; (void)ws_size;
  const float* x      = (const float*)d_in[0];
  const float* qkv_w  = (const float*)d_in[1];
  const float* qkv_b  = (const float*)d_in[2];
  const float* proj_w = (const float*)d_in[3];
  const float* proj_b = (const float*)d_in[4];
  float* out = (float*)d_out;

  char* ws = (char*)d_ws;
  size_t off = 0;
  auto alloc = [&](size_t bytes) -> void* {
    void* p = ws + off;
    off = (off + bytes + 255) & ~(size_t)255;
    return p;
  };
  unsigned short* xb   = (unsigned short*)alloc((size_t)N_TOK * C_DIM * 2);
  unsigned short* wqb  = (unsigned short*)alloc((size_t)3 * C_DIM * C_DIM * 2);
  unsigned short* wpb  = (unsigned short*)alloc((size_t)C_DIM * C_DIM * 2);
  unsigned short* qb   = (unsigned short*)alloc((size_t)HEADS * N_TOK * HD * 2);
  unsigned short* kbuf = (unsigned short*)alloc((size_t)HEADS * N_TOK * HD * 2);
  unsigned short* vtb  = (unsigned short*)alloc((size_t)HEADS * HD * N_TOK * 2 + 256);
  unsigned short* abuf = (unsigned short*)alloc((size_t)N_TOK * C_DIM * 2);

  int n1 = N_TOK * C_DIM;
  int n2 = 3 * C_DIM * C_DIM;
  int n3 = C_DIM * C_DIM;
  cvt_f32_bf16<<<(n1 + 255) / 256, 256, 0, stream>>>(x, xb, n1);
  cvt_f32_bf16<<<(n2 + 255) / 256, 256, 0, stream>>>(qkv_w, wqb, n2);
  cvt_f32_bf16<<<(n3 + 255) / 256, 256, 0, stream>>>(proj_w, wpb, n3);

  dim3 gA(N_TOK / 32 / 4, (3 * C_DIM) / 64);       // 32 x 36
  qkv_gemm<<<gA, 128, 0, stream>>>(xb, wqb, qkv_b, qb, kbuf, vtb);

  int nwaves = HEADS * (N_TOK / 16);               // 3072 waves
  local_attn<<<nwaves / 4, 128, 0, stream>>>(qb, kbuf, vtb, abuf);

  dim3 gC(N_TOK / 32 / 4, C_DIM / 64);             // 32 x 12
  proj_gemm<<<gC, 128, 0, stream>>>(abuf, wpb, proj_b, out);
}
